// KAN_44727789421160
// MI455X (gfx1250) — matrix operand order repaired and checked by reading, not compiled, not run on hardware
//
#include <hip/hip_runtime.h>

// ---------------------------------------------------------------------------
// KAN forward for MI455X (gfx1250, wave32).
//   Stage 1: pre[n,o,q] = sum_{p,h} tanh(x[n,p]*W1+b1)*W2 + sum_p b2
//            -> v_wmma_f32_16x16x32_f16, A = tanh activations (16 samples x 32 K),
//               B = W2 slice replicated across the 16 columns, C = sum_p b2.
//   Stage 2: out[n,o] = sum_q ( sum_h tanh(pre*V1+c1)*V2 + c2 )
//            -> VALU v_tanh_f32 + v_fma_f32, rebalanced across the wave via LDS.
// All weights staged into LDS (broadcast ds reads). One wave = 16 samples.
// ---------------------------------------------------------------------------

typedef __attribute__((ext_vector_type(16))) _Float16 v16h;
typedef __attribute__((ext_vector_type(8)))  float    v8f;

#define N_IN   8
#define N_OUT  4
#define N_HID  17
#define HDIM   16
#define NSAMP  32768
#define THREADS 256
#define WAVES   8      // waves per block
#define TILE    16     // samples per wave
#define NBLOCKS (NSAMP / (WAVES * TILE))   // 256

// LDS layout (float offsets)
#define W1_F   (N_OUT * N_HID * N_IN * HDIM)   // 8704
#define V1_F   (N_OUT * N_HID * HDIM)          // 1088
#define C2_F   (N_OUT * N_HID)                 // 68
#define PRE_F  (N_HID * TILE)                  // 272 floats per wave

#define OFF_W1   0
#define OFF_B1   (OFF_W1 + W1_F)               // 8704
#define OFF_V1   (OFF_B1 + W1_F)               // 17408
#define OFF_C1   (OFF_V1 + V1_F)               // 18496
#define OFF_V2   (OFF_C1 + V1_F)               // 19584
#define OFF_C2   (OFF_V2 + V1_F)               // 20672
#define OFF_B2S  (OFF_C2 + C2_F)               // 20740
#define OFF_PRE  (OFF_B2S + C2_F)              // 20808 (mult of 8 -> 32B aligned)
#define OFF_W2H  (OFF_PRE + WAVES * PRE_F)     // 22984 floats; byte 91936 (32B aligned)
#define SMEM_BYTES (OFF_W2H * 4 + W1_F * 2)    // 91936 + 17408 = 109344

__device__ __forceinline__ float fast_tanh(float x) {
#if __has_builtin(__builtin_amdgcn_tanhf)
  return __builtin_amdgcn_tanhf(x);          // v_tanh_f32 (gfx1250 TRANS op)
#else
  float y;
  asm("v_tanh_f32 %0, %1" : "=v"(y) : "v"(x));
  return y;
#endif
}

__device__ __forceinline__ void wait_lds() {
#if __has_builtin(__builtin_amdgcn_s_wait_dscnt)
  __builtin_amdgcn_s_wait_dscnt(0);
#else
  asm volatile("s_wait_dscnt 0" ::: "memory");
#endif
  asm volatile("" ::: "memory");   // compiler ordering fence
}

__global__ __launch_bounds__(THREADS, 1)
void kan_kernel(const float* __restrict__ x,
                const float* __restrict__ W1, const float* __restrict__ b1,
                const float* __restrict__ W2, const float* __restrict__ b2,
                const float* __restrict__ V1, const float* __restrict__ c1,
                const float* __restrict__ V2, const float* __restrict__ c2,
                float* __restrict__ out) {
  extern __shared__ float smem[];
  float*    sW1  = smem + OFF_W1;
  float*    sB1  = smem + OFF_B1;
  float*    sV1  = smem + OFF_V1;
  float*    sC1  = smem + OFF_C1;
  float*    sV2  = smem + OFF_V2;
  float*    sC2  = smem + OFF_C2;
  float*    sB2S = smem + OFF_B2S;
  float*    sPRE = smem + OFF_PRE;
  _Float16* sW2h = (_Float16*)(smem + OFF_W2H);

  const int tid = threadIdx.x;

  // ---- cooperative weight staging into LDS -------------------------------
  for (int i = tid; i < W1_F; i += THREADS) {
    sW1[i]  = W1[i];
    sB1[i]  = b1[i];
    sW2h[i] = (_Float16)W2[i];          // f16 B operands for WMMA
  }
  for (int i = tid; i < V1_F; i += THREADS) {
    sV1[i] = V1[i];
    sC1[i] = c1[i];
    sV2[i] = V2[i];
  }
  for (int i = tid; i < C2_F; i += THREADS) {
    sC2[i] = c2[i];
    float s = 0.f;
    #pragma unroll
    for (int p = 0; p < N_IN; ++p) s += b2[i * N_IN + p];
    sB2S[i] = s;                         // sum_p b2[o,q,p]
  }
  __syncthreads();

  // ---- per-wave setup -----------------------------------------------------
  const int lane  = tid & 31;
  const int wid   = tid >> 5;
  const int khalf = lane >> 4;           // 0: lanes 0-15, 1: lanes 16-31
  const int kofs  = khalf << 3;          // h offset 0 or 8 for A fragment
  const int m     = lane & 15;           // sample-in-tile this lane feeds
  const int tileBase = (blockIdx.x * WAVES + wid) * TILE;
  float* preW = sPRE + wid * PRE_F;

  // x row for this lane's sample (8 floats, kept in VGPRs)
  const float4* xv = (const float4*)(x + (size_t)(tileBase + m) * N_IN);
  const float4 xa = xv[0], xb = xv[1];
  float xr[N_IN] = {xa.x, xa.y, xa.z, xa.w, xb.x, xb.y, xb.z, xb.w};

  for (int o = 0; o < N_OUT; ++o) {
    // ============ stage 1: WMMA over (p,h) contraction ====================
    for (int q = 0; q < N_HID; ++q) {
      const int oq    = o * N_HID + q;
      const int wbase = oq * N_IN;       // row block in [o][q][p][h] tensors

      // C starts as broadcast of sum_p b2[o,q,p]
      const float bs = sB2S[oq];
      v8f c;
      #pragma unroll
      for (int e = 0; e < 8; ++e) c[e] = bs;

      #pragma unroll
      for (int j = 0; j < 4; ++j) {      // K = 128 = 4 chunks of 32 (p pairs)
        const int p0 = 2 * j, p1 = 2 * j + 1;
        // broadcast weight reads (2 distinct addresses per wave half)
        const v8f w0 = *(const v8f*)(sW1 + ((wbase + p0) << 4) + kofs);
        const v8f g0 = *(const v8f*)(sB1 + ((wbase + p0) << 4) + kofs);
        const v8f w1 = *(const v8f*)(sW1 + ((wbase + p1) << 4) + kofs);
        const v8f g1 = *(const v8f*)(sB1 + ((wbase + p1) << 4) + kofs);
        const float xp0 = xr[p0], xp1 = xr[p1];

        // A fragment: VGPR v(0..3) = t(p0, kofs+2v..), VGPR v(4..7) = t(p1, ..)
        v16h a;
        #pragma unroll
        for (int e = 0; e < 8; ++e) {
          a[e]     = (_Float16)fast_tanh(fmaf(xp0, w0[e], g0[e]));
          a[8 + e] = (_Float16)fast_tanh(fmaf(xp1, w1[e], g1[e]));
        }

        // B fragment: elems 0..7 = W2[o,q,p0,kofs..kofs+7], elems 8..15 = W2[o,q,p1,kofs..kofs+7] f16, replicated over columns.
        v16h bfrag;
        #pragma unroll
        for (int e = 0; e < 8; ++e) {
          bfrag[e]     = sW2h[((wbase + p0) << 4) + kofs + e];
          bfrag[8 + e] = sW2h[((wbase + p1) << 4) + kofs + e];
        }

        c = __builtin_amdgcn_wmma_f32_16x16x32_f16(
                false, a, false, bfrag, (short)0, c, false, false);
      }

      // D columns are identical: lane 0 holds pre[0..7], lane 16 pre[8..15]
      if ((lane & 15) == 0)
        *(v8f*)(preW + q * TILE + (khalf << 3)) = c;
    }
    wait_lds();

    // ============ stage 2: output MLP Phi_{o,q}, rebalanced ===============
    // two lanes per sample: lanes 0-15 handle q=0..8, lanes 16-31 q=9..16
    const int q0 = khalf ? 9 : 0;
    const int q1 = khalf ? N_HID : 9;
    float acc = 0.f;
    for (int q = q0; q < q1; ++q) {
      const float pre  = preW[q * TILE + m];
      const int   base = (o * N_HID + q) * HDIM;
      float t = sC2[o * N_HID + q];
      #pragma unroll
      for (int h = 0; h < HDIM; ++h)
        t = fmaf(fast_tanh(fmaf(pre, sV1[base + h], sC1[base + h])),
                 sV2[base + h], t);
      acc += t;
    }
    // combine the two q-halves through LDS
    preW[lane] = acc;
    wait_lds();
    if (lane < 16) {
      const float r = preW[lane] + preW[lane + 16];
      out[(size_t)(tileBase + m) * N_OUT + o] = r;
    }
    wait_lds();   // keep next o-iteration's pre stores ordered after reads
  }
}

extern "C" void kernel_launch(void* const* d_in, const int* in_sizes, int n_in,
                              void* d_out, int out_size, void* d_ws, size_t ws_size,
                              hipStream_t stream) {
  (void)in_sizes; (void)n_in; (void)out_size; (void)d_ws; (void)ws_size;
  const float* x  = (const float*)d_in[0];
  const float* W1 = (const float*)d_in[1];
  const float* b1 = (const float*)d_in[2];
  const float* W2 = (const float*)d_in[3];
  const float* b2 = (const float*)d_in[4];
  const float* V1 = (const float*)d_in[5];
  const float* c1 = (const float*)d_in[6];
  const float* V2 = (const float*)d_in[7];
  const float* c2 = (const float*)d_in[8];
  float* out = (float*)d_out;

  // allow >64KB dynamic LDS (WGP has 320KB); harmless if already set
  (void)hipFuncSetAttribute((const void*)kan_kernel,
                            hipFuncAttributeMaxDynamicSharedMemorySize,
                            (int)SMEM_BYTES);

  kan_kernel<<<NBLOCKS, THREADS, SMEM_BYTES, stream>>>(
      x, W1, b1, W2, b2, V1, c1, V2, c2, out);
}
